// TrainModel_13013750907071
// MI455X (gfx1250) — matrix operand order repaired and checked by reading, compiled, not run on hardware
//
#include <hip/hip_runtime.h>
#include <hip/hip_bf16.h>
#include <math.h>

// ---------------------------------------------------------------------------
// 2-layer LSTM + linear head + softmax/CE for MI455X (gfx1250, wave32, WMMA).
//   B=64, T=128, D=2048, H=512, C=101
//   (1) Hoisted time-parallel input projection X@W1x (+b1) -> XW1 [B*T,4H]
//       via bf16 WMMA GEMM (f32 loads, cvt_pk_bf16 in regs, f32 accum).
//   (2) Sequential T loop: two small WMMA GEMMs (M=64) + elementwise LSTM
//       per step; recurrent weights (12MB) are L2-resident (192MB L2).
//   (3) Head: per-row block computes 101 logits from LDS-staged h, softmax,
//       CE atomicAdd into cost slot.
// GEMM: one 16x64 output strip per wave (4x v_wmma_f32_16x16x32_bf16 per
// K-block, A-fragment reused 4x), compile-time bias/addend epilogues.
// ---------------------------------------------------------------------------

typedef __attribute__((ext_vector_type(16))) __bf16 v16bf;
typedef __attribute__((ext_vector_type(8)))  float  v8f;

static constexpr int B_ = 64, T_ = 128, D_ = 2048, H_ = 512, C_ = 101;
static constexpr int G_ = 4 * H_;  // 2048 gate width

#if __has_builtin(__builtin_amdgcn_tanhf)
__device__ __forceinline__ float fast_tanh(float x) { return __builtin_amdgcn_tanhf(x); }
#else
__device__ __forceinline__ float fast_tanh(float x) { return tanhf(x); }
#endif

// ---- A fragment: 16x32 bf16, rows M from A[row_base..+15], K = kb..kb+31
__device__ __forceinline__ v16bf load_a_frag(const float* __restrict__ A, int lda,
                                             int row_base, int kb, int lane) {
  const int half = lane >> 4;                 // 0: K {0..7,16..23}, 1: {8..15,24..31}
  const int m = row_base + (lane & 15);
  const float* p = A + (size_t)m * lda + kb + half * 8;
  v16bf a;
#pragma unroll
  for (int e = 0; e < 8; ++e) a[e] = (__bf16)p[e];
  const float* q = p + 16;
#pragma unroll
  for (int e = 0; e < 8; ++e) a[8 + e] = (__bf16)q[e];
  return a;
}

// ---- B fragment: 32x16 bf16, lane holds column N=col_base+(lane&15), K as A
__device__ __forceinline__ v16bf load_b_frag(const float* __restrict__ Bm, int ldb,
                                             int kb, int col_base, int lane) {
  const int half = lane >> 4;
  const float* p = Bm + (size_t)(kb + half * 8) * ldb + col_base + (lane & 15);
  v16bf b;
#pragma unroll
  for (int e = 0; e < 8; ++e) b[e] = (__bf16)p[(size_t)e * ldb];
#pragma unroll
  for (int e = 0; e < 8; ++e) b[8 + e] = (__bf16)p[(size_t)(16 + e) * ldb];
  return b;
}

// ---- Tiled GEMM: Out[M,N] = A[M,K]@Bm[K,N] (+bias[N]) (+addend[row,:])
// Each wave computes a 16 x (16*NT) strip: A fragment reused across NT WMMAs.
template <int NT, bool HAS_BIAS, bool HAS_ADDEND, bool PREFETCH>
__global__ __launch_bounds__(256) void gemm_wmma(
    const float* __restrict__ A, int lda,
    const float* __restrict__ Bm, int ldb,
    float* __restrict__ Out, int ldo,
    const float* __restrict__ bias,
    const float* __restrict__ addend, int add_stride,
    int Mtiles, int Ngroups, int K) {
  const int wave = (blockIdx.x * blockDim.x + threadIdx.x) >> 5;
  const int lane = threadIdx.x & 31;
  if (wave >= Mtiles * Ngroups) return;       // whole-wave exit only
  const int tm = wave / Ngroups;
  const int tg = wave - tm * Ngroups;
  const int row_base = tm * 16;
  const int col_base = tg * 16 * NT;

  v8f acc[NT];
#pragma unroll
  for (int j = 0; j < NT; ++j) acc[j] = v8f{};

  for (int kb = 0; kb < K; kb += 32) {
    if (PREFETCH && (kb + 64 < K)) {
      // warm the next-next B panel line for this lane (global_prefetch_b8)
      __builtin_prefetch(Bm + (size_t)(kb + 64 + lane) * ldb + col_base, 0, 3);
    }
    const v16bf a = load_a_frag(A, lda, row_base, kb, lane);
    v16bf b[NT];
#pragma unroll
    for (int j = 0; j < NT; ++j)
      b[j] = load_b_frag(Bm, ldb, kb, col_base + 16 * j, lane);
#pragma unroll
    for (int j = 0; j < NT; ++j)
      acc[j] = __builtin_amdgcn_wmma_f32_16x16x32_bf16(
          /*neg_a=*/false, a, /*neg_b=*/false, b[j],
          /*c_mod=*/(short)0, acc[j], /*reuse_a=*/false, /*reuse_b=*/false);
  }

  // D layout: VGPR i -> M = i (+8 for upper half-wave), N = lane&15
  const int lane15 = lane & 15;
  const int rbase = row_base + ((lane >> 4) << 3);
#pragma unroll
  for (int j = 0; j < NT; ++j) {
    const int col = col_base + 16 * j + lane15;
    const float bv = HAS_BIAS ? bias[col] : 0.0f;
#pragma unroll
    for (int i = 0; i < 8; ++i) {
      const int row = rbase + i;
      float v = acc[j][i] + bv;
      if (HAS_ADDEND) v += addend[(size_t)row * (size_t)add_stride + col];
      Out[(size_t)row * ldo + col] = v;
    }
  }
}

// ---- Elementwise LSTM cell update. gates [64, 2048] order: i, j, f, o.
__global__ __launch_bounds__(256) void lstm_elem(
    const float* __restrict__ gates,
    float* __restrict__ c,            // [64, 512]
    float* __restrict__ hcat,         // [64, 1024]
    int hcat_col_off,
    float* __restrict__ h_out,        // optional: h2_all + t*H (row stride T*H)
    int h_out_stride) {
  const int idx = blockIdx.x * blockDim.x + threadIdx.x;
  if (idx >= B_ * H_) return;
  const int b = idx >> 9;             // /512
  const int h = idx & (H_ - 1);
  const float* g = gates + (size_t)b * G_;
  const float gi = g[h];
  const float gj = g[H_ + h];
  const float gf = g[2 * H_ + h];
  const float go = g[3 * H_ + h];
  const float cv = c[idx];
  const float fg = 1.0f / (1.0f + __expf(-(gf + 1.0f)));   // forget bias = 1.0
  const float ig = 1.0f / (1.0f + __expf(-gi));
  const float og = 1.0f / (1.0f + __expf(-go));
  const float nc = cv * fg + ig * fast_tanh(gj);
  const float nh = fast_tanh(nc) * og;
  c[idx] = nc;
  hcat[b * 2 * H_ + hcat_col_off + h] = nh;
  if (h_out) h_out[(size_t)b * h_out_stride + h] = nh;
}

// ---- Zero c1,c2,hcat (contiguous) + cost accumulator in d_out
__global__ __launch_bounds__(256) void init_state(float* __restrict__ p, int n,
                                                  float* __restrict__ cost) {
  const int i = blockIdx.x * blockDim.x + threadIdx.x;
  if (i < n) p[i] = 0.0f;
  if (i == 0) *cost = 0.0f;
}

// ---- Head: logits -> softmax preds + CE accumulation. One block per row.
__global__ __launch_bounds__(128) void logits_softmax_ce(
    const float* __restrict__ h2_all,   // [B*T, H]
    const float* __restrict__ Wout,     // [H, C]
    const float* __restrict__ bout,     // [C]
    const int* __restrict__ labels,     // [B*T]
    float* __restrict__ preds,          // [B*T, C]
    float* __restrict__ cost_acc) {
  __shared__ float hrow[H_];
  __shared__ float logit[C_];
  __shared__ float red[128];
  const int r = blockIdx.x;
  const int tid = threadIdx.x;

  const float* h = h2_all + (size_t)r * H_;
  for (int i = tid; i < H_; i += 128) hrow[i] = h[i];
  __syncthreads();

  for (int j = tid; j < C_; j += 128) {
    float s = bout[j];
    for (int k = 0; k < H_; ++k) s = fmaf(hrow[k], Wout[k * C_ + j], s);
    logit[j] = s;
  }
  __syncthreads();

  float m = -INFINITY;
  for (int j = tid; j < C_; j += 128) m = fmaxf(m, logit[j]);
  red[tid] = m;
  __syncthreads();
  for (int s = 64; s > 0; s >>= 1) {
    if (tid < s) red[tid] = fmaxf(red[tid], red[tid + s]);
    __syncthreads();
  }
  m = red[0];
  __syncthreads();

  float sum = 0.0f;
  for (int j = tid; j < C_; j += 128) sum += __expf(logit[j] - m);
  red[tid] = sum;
  __syncthreads();
  for (int s = 64; s > 0; s >>= 1) {
    if (tid < s) red[tid] += red[tid + s];
    __syncthreads();
  }
  sum = red[0];

  const float inv = 1.0f / sum;
  for (int j = tid; j < C_; j += 128)
    preds[(size_t)r * C_ + j] = __expf(logit[j] - m) * inv;

  if (tid == 0) {
    const int lab = labels[r];
    const float ce = (m + __logf(sum)) - logit[lab];
    atomicAdd(cost_acc, ce * (1.0f / (float)B_));   // mean over B of sum over T
  }
}

// ---------------------------------------------------------------------------
extern "C" void kernel_launch(void* const* d_in, const int* in_sizes, int n_in,
                              void* d_out, int out_size, void* d_ws, size_t ws_size,
                              hipStream_t stream) {
  (void)in_sizes; (void)n_in; (void)out_size; (void)ws_size;
  const float* inputs = (const float*)d_in[0];   // [B,T,D]
  const int*   labels = (const int*)d_in[1];     // [B,T]
  const float* W1     = (const float*)d_in[2];   // [D+H, 4H]
  const float* b1     = (const float*)d_in[3];   // [4H]
  const float* W2     = (const float*)d_in[4];   // [2H, 4H]
  const float* b2     = (const float*)d_in[5];   // [4H]
  const float* Wout   = (const float*)d_in[6];   // [H, C]
  const float* bout   = (const float*)d_in[7];   // [C]
  float* out = (float*)d_out;                    // preds [B,T,C] ++ cost [1]

  // Workspace layout (floats): ~85 MB total
  float* XW1    = (float*)d_ws;                        // [B*T, 4H]  64 MB
  float* gates  = XW1   + (size_t)B_ * T_ * G_;        // [B, 4H]
  float* c1     = gates + (size_t)B_ * G_;             // [B, H]
  float* c2     = c1    + (size_t)B_ * H_;             // [B, H]
  float* hcat   = c2    + (size_t)B_ * H_;             // [B, 2H] = [h1 | h2]
  float* h2_all = hcat  + (size_t)B_ * 2 * H_;         // [B, T, H] 16 MB

  float* cost_slot = out + (size_t)B_ * T_ * C_;

  // (0) zero c1, c2, hcat (contiguous) + cost slot
  const int nzero = B_ * H_ * 2 + B_ * 2 * H_;
  init_state<<<(nzero + 255) / 256, 256, 0, stream>>>(c1, nzero, cost_slot);

  // (1) XW1 = inputs(flat [B*T, D]) @ W1[0:D,:] + b1
  //     Mtiles=512, Ngroups=32 (NT=4) -> 16384 waves -> 2048 blocks
  gemm_wmma<4, true, false, true><<<2048, 256, 0, stream>>>(
      inputs, D_, W1, G_, XW1, G_, b1,
      nullptr, 0, (B_ * T_) / 16, G_ / 64, D_);

  // (2) sequential recurrence
  const float* W1h = W1 + (size_t)D_ * G_;             // rows D..D+H-1
  for (int t = 0; t < T_; ++t) {
    // gates1 = h1 @ W1h + XW1[:, t, :]   (Mtiles=4, Ngroups=32 -> 128 waves)
    gemm_wmma<4, false, true, false><<<16, 256, 0, stream>>>(
        hcat, 2 * H_, W1h, G_, gates, G_,
        nullptr, XW1 + (size_t)t * G_, T_ * G_,
        B_ / 16, G_ / 64, H_);
    lstm_elem<<<(B_ * H_) / 256, 256, 0, stream>>>(
        gates, c1, hcat, /*col_off=*/0, nullptr, 0);

    // gates2 = [h1_new | h2_prev] @ W2 + b2
    gemm_wmma<4, true, false, false><<<16, 256, 0, stream>>>(
        hcat, 2 * H_, W2, G_, gates, G_,
        b2, nullptr, 0, B_ / 16, G_ / 64, 2 * H_);
    lstm_elem<<<(B_ * H_) / 256, 256, 0, stream>>>(
        gates, c2, hcat, /*col_off=*/H_, h2_all + (size_t)t * H_, T_ * H_);
  }

  // (3) head: softmax preds + CE cost
  logits_softmax_ce<<<B_ * T_, 128, 0, stream>>>(
      h2_all, Wout, bout, labels, out, cost_slot);
}
